// EGNNLayer_38938173506036
// MI455X (gfx1250) — matrix operand order repaired and checked by reading, not compiled, not run on hardware
//
#include <hip/hip_runtime.h>
#include <hip/hip_bf16.h>
#include <stdint.h>

// EGNN layer, fused for MI455X (gfx1250, wave32, WMMA bf16 path).
// N=50000 nodes, E=800000 edges, F=H=128.
//
// Roofline: edge MLP ~79 GFLOP dominates; x/pos/weights/accumulators all fit in
// 192MB L2, so the fused edge kernel is matrix-core bound -> bf16 WMMA with f32
// accumulate. Weights are pre-swizzled into per-lane fragment order so each
// B-fragment is one aligned 32B load per lane (coalesced 1KB per wave).

#define N_NODES 50000
#define N_EDGES 800000
#define FDIM 128
#define HDIM 128

typedef __attribute__((ext_vector_type(16))) __bf16 v16bf;
typedef __attribute__((ext_vector_type(8)))  __bf16 v8bf;
typedef __attribute__((ext_vector_type(4)))  __bf16 v4bf;
typedef __attribute__((ext_vector_type(8)))  float  v8f;

__device__ __forceinline__ float silu_f(float v) {
    return v * (1.0f / (1.0f + __expf(-v)));
}

// ---- WMMA fragment builders (ISA 7.12.2 layouts, wave32) ----

// A: 16x32 bf16 tile from LDS. Per lane the 16 values live in two contiguous
// 16-byte runs -> exactly 2x ds_load_b128.
__device__ __forceinline__ v16bf load_A_frag(const __bf16* sA, int lda, int kc, int lane) {
    const int half = lane >> 4;
    const int m    = lane & 15;
    const __bf16* rowp = sA + m * lda;
    v8bf lo = *(const v8bf*)(rowp + kc + half * 8);
    v8bf hi = *(const v8bf*)(rowp + kc + 16 + half * 8);
    return __builtin_shufflevector(lo, hi, 0, 1, 2, 3, 4, 5, 6, 7,
                                   8, 9, 10, 11, 12, 13, 14, 15);
}

// B: pre-swizzled fragment layout. Fragment (cbi,kci) is 1KB: lane l's 16
// bf16 at offset l*16. One aligned 32B load per lane -> 2x global_load_b128.
__device__ __forceinline__ v16bf load_B_swz(const __bf16* __restrict__ Wswz, int nkc,
                                            int kci, int cbi, int lane) {
    return *(const v16bf*)(Wswz + (((size_t)(cbi * nkc + kci)) << 9) + lane * 16);
}

// ---- weight swizzle: W[K][128] f32 row-major -> bf16 fragment order ----
// out[((cbi*nkc + kci)*512) + l*16 + j] = W[(kci*32 + (j>>3)*16 + (l>>4)*8 + (j&7))*128 + cbi*16 + (l&15)]
__global__ void swizzle_weights(const float* __restrict__ W, int nkc,
                                __bf16* __restrict__ out) {
    int i = blockIdx.x * blockDim.x + threadIdx.x;
    int total = 8 * nkc * 512;
    if (i >= total) return;
    int frag   = i >> 9;
    int within = i & 511;
    int cbi = frag / nkc;
    int kci = frag % nkc;
    int l = within >> 4;
    int j = within & 15;
    int k = kci * 32 + (j >> 3) * 16 + (l >> 4) * 8 + (j & 7);
    int n = cbi * 16 + (l & 15);
    out[i] = (__bf16)W[k * 128 + n];
}

__global__ void zero_ws(float* __restrict__ p, int n) {
    int i = blockIdx.x * blockDim.x + threadIdx.x;
    if (i < n) p[i] = 0.0f;
}

// ---- fused edge kernel ----
// block = 256 threads = 8 waves; 64-edge tile (4 M-subtiles of 16).
// Each wave owns 16 output cols; B fragments are loaded once and feed 4 WMMAs.
__global__ __launch_bounds__(256)
void egnn_edge_kernel(const float* __restrict__ x, const float* __restrict__ pos,
                      const float* __restrict__ We1,   // f32: row 256 (dij column)
                      const float* __restrict__ be1, const float* __restrict__ be2,
                      const float* __restrict__ Wc,  const float* __restrict__ bc,
                      const long long* __restrict__ ei,
                      const __bf16* __restrict__ We1s, const __bf16* __restrict__ We2s,
                      float* __restrict__ agg, float* __restrict__ delta) {
    __shared__ __align__(16) __bf16 sA[64 * 256];   // [x[row] || x[col]]
    __shared__ __align__(16) __bf16 sH[64 * 128];   // silu(layer1)
    __shared__ __align__(16) __bf16 sM[64 * 128];   // m_ij
    __shared__ int   sRow[64];
    __shared__ int   sCol[64];
    __shared__ float sRij[64][3];
    __shared__ float sDij[64];

    const int t    = threadIdx.x;
    const int lane = t & 31;
    const int wave = t >> 5;          // 0..7 -> output col block
    const int e0   = blockIdx.x * 64;

    if (t < 64) {
        int e = e0 + t;
        int r = (int)ei[e];
        int c = (int)ei[N_EDGES + e];
        sRow[t] = r;
        sCol[t] = c;
        float d = 0.0f;
#pragma unroll
        for (int k = 0; k < 3; ++k) {
            float rv = pos[r * 3 + k] - pos[c * 3 + k];
            sRij[t][k] = rv;
            d += rv * rv;
        }
        sDij[t] = d;
    }
    __syncthreads();

    // Gather 64 x 256 features as float4 (rows of x are contiguous; hits L2),
    // convert to bf16 and store packed 8B chunks to LDS.
#pragma unroll
    for (int it = 0; it < 16; ++it) {
        int q  = t + it * 256;        // float4 index 0..4095
        int e  = q >> 6;              // 64 float4 per 256-col row
        int f4 = (q & 63) << 2;
        int node = (f4 < 128) ? sRow[e] : sCol[e];
        int fc   = f4 & 127;
        const float4 v = *(const float4*)(x + (size_t)node * 128 + fc);
        v4bf h = { (__bf16)v.x, (__bf16)v.y, (__bf16)v.z, (__bf16)v.w };
        *(v4bf*)(sA + e * 256 + f4) = h;
    }
    __syncthreads();

    const int n    = wave * 16 + (lane & 15);
    const int half = lane >> 4;

    // GEMM1: [64,256] x [256,16] per wave; 8 K-chunks x 4 M-subtiles = 32 WMMAs.
    v8f acc[4] = {};
#pragma unroll
    for (int kci = 0; kci < 8; ++kci) {
        v16bf b = load_B_swz(We1s, 8, kci, wave, lane);
#pragma unroll
        for (int mt = 0; mt < 4; ++mt) {
            v16bf a = load_A_frag(sA + mt * 16 * 256, 256, kci * 32, lane);
            acc[mt] = __builtin_amdgcn_wmma_f32_16x16x32_bf16(false, a, false, b,
                                                              (short)0, acc[mt],
                                                              false, false);
        }
    }
    {
        const float bias  = be1[n];
        const float wlast = We1[256 * 128 + n];   // dij column as rank-1 epilogue
#pragma unroll
        for (int mt = 0; mt < 4; ++mt)
#pragma unroll
            for (int v = 0; v < 8; ++v) {
                int m = mt * 16 + v + 8 * half;
                float val = acc[mt][v] + bias + sDij[m] * wlast;
                sH[m * 128 + n] = (__bf16)silu_f(val);
            }
    }
    __syncthreads();

    // GEMM2: [64,128] x [128,16] per wave; 4 K-chunks x 4 M-subtiles = 16 WMMAs.
    v8f acc2[4] = {};
#pragma unroll
    for (int kci = 0; kci < 4; ++kci) {
        v16bf b = load_B_swz(We2s, 4, kci, wave, lane);
#pragma unroll
        for (int mt = 0; mt < 4; ++mt) {
            v16bf a = load_A_frag(sH + mt * 16 * 128, 128, kci * 32, lane);
            acc2[mt] = __builtin_amdgcn_wmma_f32_16x16x32_bf16(false, a, false, b,
                                                               (short)0, acc2[mt],
                                                               false, false);
        }
    }
    {
        const float bias = be2[n];
#pragma unroll
        for (int mt = 0; mt < 4; ++mt)
#pragma unroll
            for (int v = 0; v < 8; ++v) {
                int m  = mt * 16 + v + 8 * half;
                float mv = silu_f(acc2[mt][v] + bias);
                sM[m * 128 + n] = (__bf16)mv;
                unsafeAtomicAdd(&agg[(size_t)sRow[m] * 128 + n], mv);
            }
    }
    __syncthreads();

    // w_ij = silu(m_ij . Wc + bc); delta[row] += rij * w_ij
    if (t < 64) {
        float aw = bc[0];
#pragma unroll 8
        for (int k = 0; k < 128; ++k) aw += (float)sM[t * 128 + k] * Wc[k];
        float w = silu_f(aw);
        int r = sRow[t];
#pragma unroll
        for (int k = 0; k < 3; ++k)
            unsafeAtomicAdd(&delta[(size_t)r * 3 + k], sRij[t][k] * w);
    }
}

// ---- node kernel: x_new = silu([x||agg] @ Wn1 + bn1) @ Wn2 + bn2 ----
// 16-node tiles (3125 blocks exactly); only ~5 GFLOP so simple structure is fine.
__global__ __launch_bounds__(256)
void egnn_node_kernel(const float* __restrict__ x, const float* __restrict__ agg,
                      const float* __restrict__ bn1, const float* __restrict__ bn2,
                      const __bf16* __restrict__ Wn1s, const __bf16* __restrict__ Wn2s,
                      float* __restrict__ xout) {
    __shared__ __align__(16) __bf16 sA[16 * 256];
    __shared__ __align__(16) __bf16 sH[16 * 128];

    const int t    = threadIdx.x;
    const int lane = t & 31;
    const int wave = t >> 5;
    const int n0   = blockIdx.x * 16;

#pragma unroll
    for (int it = 0; it < 4; ++it) {
        int q  = t + it * 256;        // float4 index 0..1023
        int r  = q >> 6;
        int f4 = (q & 63) << 2;
        int node = n0 + r;
        const float* src = (f4 < 128) ? (x + (size_t)node * 128 + f4)
                                      : (agg + (size_t)node * 128 + (f4 - 128));
        const float4 v = *(const float4*)src;
        v4bf h = { (__bf16)v.x, (__bf16)v.y, (__bf16)v.z, (__bf16)v.w };
        *(v4bf*)(sA + r * 256 + f4) = h;
    }
    __syncthreads();

    const int n    = wave * 16 + (lane & 15);
    const int half = lane >> 4;

    v8f acc = {};
#pragma unroll
    for (int kci = 0; kci < 8; ++kci) {
        v16bf a = load_A_frag(sA, 256, kci * 32, lane);
        v16bf b = load_B_swz(Wn1s, 8, kci, wave, lane);
        acc = __builtin_amdgcn_wmma_f32_16x16x32_bf16(false, a, false, b, (short)0, acc,
                                                      false, false);
    }
    {
        const float bias = bn1[n];
#pragma unroll
        for (int v = 0; v < 8; ++v) {
            int m = v + 8 * half;
            sH[m * 128 + n] = (__bf16)silu_f(acc[v] + bias);
        }
    }
    __syncthreads();

    v8f acc2 = {};
#pragma unroll
    for (int kci = 0; kci < 4; ++kci) {
        v16bf a = load_A_frag(sH, 128, kci * 32, lane);
        v16bf b = load_B_swz(Wn2s, 4, kci, wave, lane);
        acc2 = __builtin_amdgcn_wmma_f32_16x16x32_bf16(false, a, false, b, (short)0, acc2,
                                                       false, false);
    }
    {
        const float bias = bn2[n];
#pragma unroll
        for (int v = 0; v < 8; ++v) {
            int m = v + 8 * half;
            xout[(size_t)(n0 + m) * 128 + n] = acc2[v] + bias;   // no activation
        }
    }
}

__global__ void pos_update(const float* __restrict__ pos, const float* __restrict__ delta,
                           float* __restrict__ pout) {
    int i = blockIdx.x * blockDim.x + threadIdx.x;
    if (i < N_NODES * 3) pout[i] = pos[i] + 0.01f * delta[i];
}

extern "C" void kernel_launch(void* const* d_in, const int* in_sizes, int n_in,
                              void* d_out, int out_size, void* d_ws, size_t ws_size,
                              hipStream_t stream) {
    const float* x   = (const float*)d_in[0];
    const float* pos = (const float*)d_in[1];
    const float* We1 = (const float*)d_in[2];
    const float* be1 = (const float*)d_in[3];
    const float* We2 = (const float*)d_in[4];
    const float* be2 = (const float*)d_in[5];
    const float* Wn1 = (const float*)d_in[6];
    const float* bn1 = (const float*)d_in[7];
    const float* Wn2 = (const float*)d_in[8];
    const float* bn2 = (const float*)d_in[9];
    const float* Wc  = (const float*)d_in[10];
    const float* bc  = (const float*)d_in[11];
    const long long* ei = (const long long*)d_in[12];   // int64 edge_index [2,E]

    // workspace: agg [N,128] f32 | delta [N,3] f32 | swizzled bf16 weights
    float*  agg   = (float*)d_ws;
    float*  delta = agg + (size_t)N_NODES * HDIM;
    __bf16* wb    = (__bf16*)(delta + (size_t)N_NODES * 3);   // 32B-aligned
    __bf16* We1s  = wb;                     // 8 cb * 8 kc * 512 = 32768
    __bf16* We2s  = We1s + 32768;           // 8 * 4 * 512      = 16384
    __bf16* Wn1s  = We2s + 16384;           // 32768
    __bf16* Wn2s  = Wn1s + 32768;           // 16384

    float* xout = (float*)d_out;                        // [N,128]
    float* pout = xout + (size_t)N_NODES * FDIM;        // [N,3]

    swizzle_weights<<<(32768 + 255) / 256, 256, 0, stream>>>(We1, 8, We1s);
    swizzle_weights<<<(16384 + 255) / 256, 256, 0, stream>>>(We2, 4, We2s);
    swizzle_weights<<<(32768 + 255) / 256, 256, 0, stream>>>(Wn1, 8, Wn1s);
    swizzle_weights<<<(16384 + 255) / 256, 256, 0, stream>>>(Wn2, 4, Wn2s);

    const int nzero = N_NODES * HDIM + N_NODES * 3;     // agg+delta contiguous
    zero_ws<<<(nzero + 255) / 256, 256, 0, stream>>>(agg, nzero);

    egnn_edge_kernel<<<N_EDGES / 64, 256, 0, stream>>>(x, pos, We1, be1, be2, Wc, bc, ei,
                                                       We1s, We2s, agg, delta);

    egnn_node_kernel<<<N_NODES / 16, 256, 0, stream>>>(x, agg, bn1, bn2, Wn1s, Wn2s, xout);

    pos_update<<<(N_NODES * 3 + 255) / 256, 256, 0, stream>>>(pos, delta, pout);
}
